// Speller_17300128268733
// MI455X (gfx1250) — compile-verified, matrix operand order checked by reading, not run on hardware
//
#include <hip/hip_runtime.h>
#include <hip/hip_bf16.h>

// ---------- types ----------
typedef __bf16 bf16_t;
typedef bf16_t v16bf __attribute__((ext_vector_type(16)));
typedef bf16_t v8bf  __attribute__((ext_vector_type(8)));
typedef float  v8f   __attribute__((ext_vector_type(8)));

#define B_   64
#define T_   1024
#define HC_  512
#define HS_  512
#define V_   64
#define HM_  128
#define L_   128      // MAXLEN
#define KX_  1088     // V + Hc + Hs
#define G4_  2048     // 4*Hs
#define EPS_ 1e-5f

static __device__ __forceinline__ bf16_t f2bf(float f) {
    unsigned u = __float_as_uint(f);
    unsigned r = (u + 0x7fffu + ((u >> 16) & 1u)) >> 16;
    union { unsigned short s; bf16_t b; } cv;
    cv.s = (unsigned short)r;
    return cv.b;
}
static __device__ __forceinline__ float sigmoidf_(float x) {
    return 1.0f / (1.0f + __expf(-x));
}

// Build a 16x32 A fragment (row-major bf16 source) for this lane.
// ISA 7.12.2 (16-bit A): lanes 0-15 hold K 0..7 / 16..23 in the two
// 8-element halves, lanes 16-31 hold K 8..15 / 24..31 (k0 includes lhalf*8).
static __device__ __forceinline__ v16bf load_a_frag(const bf16_t* row, int k0) {
    v8bf lo = *(const v8bf*)(row + k0);
    v8bf hi = *(const v8bf*)(row + k0 + 16);
    v16bf a;
#pragma unroll
    for (int i = 0; i < 8; ++i) { a[i] = lo[i]; a[8 + i] = hi[i]; }
    return a;
}
static __device__ __forceinline__ v16bf load_pk_frag(const bf16_t* bp) {
    v8bf lo = *(const v8bf*)bp;
    v8bf hi = *(const v8bf*)(bp + 8);
    v16bf b;
#pragma unroll
    for (int i = 0; i < 8; ++i) { b[i] = lo[i]; b[8 + i] = hi[i]; }
    return b;
}

// ---------- prep kernels ----------
__global__ void conv_h_bf16_k(const float* __restrict__ h, bf16_t* __restrict__ hbf) {
    size_t i = (size_t)blockIdx.x * blockDim.x + threadIdx.x;
    if (i < (size_t)B_ * T_ * HC_) hbf[i] = f2bf(h[i]);
}

// Pack h[b] (K=t=1024, N=c=512) into per-wave WMMA B fragments for the ctx GEMM:
// flat = (((b*32 + kt)*32 + nt)*32 + lane)*16 + j
__global__ void pack_hB_k(const float* __restrict__ h, bf16_t* __restrict__ hBpack) {
    size_t idx = (size_t)blockIdx.x * blockDim.x + threadIdx.x;
    if (idx >= (size_t)B_ * T_ * HC_) return;
    int j = idx & 15;
    int l = (idx >> 4) & 31;
    size_t tile = idx >> 9;
    int nt = (int)(tile & 31);
    int kt = (int)((tile >> 5) & 31);
    int b  = (int)(tile >> 10);
    int k = kt * 32 + ((l >> 4) & 1) * 8 + (j < 8 ? j : 8 + j);  // t index
    int n = nt * 16 + (l & 15);                                  // c index
    hBpack[idx] = f2bf(h[((size_t)b * T_ + k) * HC_ + n]);
}

// Pack Wcat = [W_ih.T ; W_hh.T]  (K=1088, N=2048) into per-wave B fragments:
// flat = ((kt*128 + nt)*32 + lane)*16 + j
__global__ void pack_gates_w_k(const float* __restrict__ W_ih, const float* __restrict__ W_hh,
                               const float* __restrict__ b_ih, const float* __restrict__ b_hh,
                               bf16_t* __restrict__ Wpack, float* __restrict__ bsum) {
    int idx = blockIdx.x * 256 + threadIdx.x;
    const int total = 34 * 128 * 512;
    if (idx < total) {
        int j = idx & 15;
        int l = (idx >> 4) & 31;
        int tile = idx >> 9;
        int nt = tile % 128, kt = tile / 128;
        int n = nt * 16 + (l & 15);
        int k = kt * 32 + ((l >> 4) & 1) * 8 + (j < 8 ? j : 8 + j);
        float v = (k < 576) ? W_ih[(size_t)n * 576 + k]
                            : W_hh[(size_t)n * 512 + (k - 576)];
        Wpack[idx] = f2bf(v);
    }
    if (idx < G4_) bsum[idx] = b_ih[idx] + b_hh[idx];
}

__global__ void pack_fc_k(const float* __restrict__ fc1_w, const float* __restrict__ fc2_w,
                          bf16_t* __restrict__ fc1p, bf16_t* __restrict__ fc2p) {
    int idx = blockIdx.x * 256 + threadIdx.x;
    const int T1 = 32 * 8 * 512;   // fc1: K=1024, N=128
    const int T2 = 4 * 4 * 512;    // fc2: K=128,  N=64
    if (idx < T1) {
        int j = idx & 15, l = (idx >> 4) & 31, tile = idx >> 9;
        int nt = tile % 8, kt = tile / 8;
        int n = nt * 16 + (l & 15);
        int k = kt * 32 + ((l >> 4) & 1) * 8 + (j < 8 ? j : 8 + j);
        fc1p[idx] = f2bf(fc1_w[(size_t)n * 1024 + k]);
    } else if (idx < T1 + T2) {
        int m = idx - T1;
        int j = m & 15, l = (m >> 4) & 31, tile = m >> 9;
        int nt = tile % 4, kt = tile / 4;
        int n = nt * 16 + (l & 15);
        int k = kt * 32 + ((l >> 4) & 1) * 8 + (j < 8 ? j : 8 + j);
        fc2p[m] = f2bf(fc2_w[(size_t)n * 128 + k]);
    }
}

__global__ void init_state_k(const float* __restrict__ h, bf16_t* __restrict__ Xbf,
                             float* __restrict__ cs, int* __restrict__ lens,
                             bf16_t* __restrict__ featbf) {
    int idx = blockIdx.x * 256 + threadIdx.x;
    if (idx < B_ * KX_) {
        int b = idx / KX_, c = idx % KX_;
        float v;
        if (c < V_)        v = (c == 0) ? 1.0f : 0.0f;             // one-hot SOS
        else if (c < 576)  v = h[(size_t)b * T_ * HC_ + (c - V_)]; // h[:,0,:]
        else               v = 0.0f;                               // hs0 = 0
        Xbf[idx] = f2bf(v);
    }
    if (idx < B_ * HS_) cs[idx] = 0.0f;
    if (idx < B_)       lens[idx] = L_;
    if (idx < B_ * 1024) featbf[idx] = f2bf(0.0f);
}

// ---------- generic WMMA GEMM, 4 N-tiles per wave (A reused 4x) ----------
__global__ void wmma_gemm_k(const bf16_t* __restrict__ A, int lda,
                            const bf16_t* __restrict__ Bpack, int ntiles, int ktiles,
                            int mtiles,
                            float* __restrict__ D, int ldd,
                            const float* __restrict__ bias,
                            bf16_t* __restrict__ Dbf, int lddbf) {
    int wave = (blockIdx.x * blockDim.x + threadIdx.x) >> 5;
    int nblocks = ntiles >> 2;
    if (wave >= mtiles * nblocks) return;
    int l = threadIdx.x & 31;
    int mt = wave / nblocks, nb = wave % nblocks;
    int lhalf = (l >> 4) & 1;
    int row = mt * 16 + (l & 15);
    v8f acc0 = {}, acc1 = {}, acc2 = {}, acc3 = {};
    for (int kt = 0; kt < ktiles; ++kt) {
        int k0 = kt * 32 + lhalf * 8;
        v16bf a = load_a_frag(A + (size_t)row * lda, k0);
        const bf16_t* bp = Bpack + (((size_t)kt * ntiles + nb * 4) * 32 + l) * 16;
        v16bf b0 = load_pk_frag(bp);
        v16bf b1 = load_pk_frag(bp + 512);
        v16bf b2 = load_pk_frag(bp + 1024);
        v16bf b3 = load_pk_frag(bp + 1536);
        acc0 = __builtin_amdgcn_wmma_f32_16x16x32_bf16(false, a, false, b0, (short)0, acc0, false, false);
        acc1 = __builtin_amdgcn_wmma_f32_16x16x32_bf16(false, a, false, b1, (short)0, acc1, false, false);
        acc2 = __builtin_amdgcn_wmma_f32_16x16x32_bf16(false, a, false, b2, (short)0, acc2, false, false);
        acc3 = __builtin_amdgcn_wmma_f32_16x16x32_bf16(false, a, false, b3, (short)0, acc3, false, false);
    }
    int rbase = mt * 16 + lhalf * 8;
#pragma unroll
    for (int q = 0; q < 4; ++q) {
        v8f acc = (q == 0) ? acc0 : (q == 1) ? acc1 : (q == 2) ? acc2 : acc3;
        int col = (nb * 4 + q) * 16 + (l & 15);
        float bv = bias ? bias[col] : 0.0f;
#pragma unroll
        for (int r = 0; r < 8; ++r) {
            float v = acc[r] + bv;
            if (D)   D[(size_t)(rbase + r) * ldd + col] = v;
            if (Dbf) Dbf[(size_t)(rbase + r) * lddbf + col] = f2bf(v);
        }
    }
}

// ---------- LSTM cell pointwise ----------
__global__ void lstm_cell_k(const float* __restrict__ gates, float* __restrict__ cs,
                            bf16_t* __restrict__ Xbf, bf16_t* __restrict__ featbf) {
    int idx = blockIdx.x * blockDim.x + threadIdx.x;
    if (idx >= B_ * HS_) return;
    int b = idx >> 9, j = idx & 511;
    const float* g = gates + (size_t)b * G4_;
    float i_ = sigmoidf_(g[j]);
    float f_ = sigmoidf_(g[512 + j]);
    float gg = tanhf(g[1024 + j]);
    float o_ = sigmoidf_(g[1536 + j]);
    float c = f_ * cs[idx] + i_ * gg;
    float h_ = o_ * tanhf(c);
    cs[idx] = c;
    bf16_t hb = f2bf(h_);
    Xbf[(size_t)b * KX_ + 576 + j] = hb;   // hs slot for next-step gates
    featbf[(size_t)b * 1024 + j] = hb;     // feat[:, :512] = hs
}

// ---------- attention scores: e[b,t] = <hs[b], h[b,t,:]> via WMMA ----------
__global__ void attn_scores_k(const bf16_t* __restrict__ hbf,
                              const bf16_t* __restrict__ featbf,  // hs bf16, row stride 1024
                              float* __restrict__ e) {
    int wave = (blockIdx.x * blockDim.x + threadIdx.x) >> 5;
    if (wave >= B_ * (T_ / 16)) return;
    int l = threadIdx.x & 31;
    int b = wave >> 6;
    int tt = wave & 63;
    int lhalf = (l >> 4) & 1;
    int trow = tt * 16 + (l & 15);
    const bf16_t* hrow = hbf + ((size_t)b * T_ + trow) * HC_;
    const bf16_t* hsrow = featbf + (size_t)b * 1024;
    v8f acc = {};
    for (int kt = 0; kt < HC_ / 32; ++kt) {
        int k0 = kt * 32 + lhalf * 8;
        v16bf a = load_a_frag(hrow, k0);
        v16bf bm = load_a_frag(hsrow, k0);   // hs replicated over all 16 N columns
        acc = __builtin_amdgcn_wmma_f32_16x16x32_bf16(false, a, false, bm,
                                                      (short)0, acc, false, false);
    }
    if ((l & 15) == 0) {     // N==0 column; lane 0 -> M 0..7, lane 16 -> M 8..15
        int mbase = tt * 16 + lhalf * 8;
#pragma unroll
        for (int r = 0; r < 8; ++r)
            e[(size_t)b * T_ + mbase + r] = acc[r];
    }
}

// ---------- global max over e[64,1024] (matches jnp.max(e)) ----------
__global__ void reduce_max_k(const float* __restrict__ e, float* __restrict__ emax) {
    __shared__ float sm[1024];
    float m = -3.4e38f;
    for (int i = threadIdx.x; i < B_ * T_; i += 1024) m = fmaxf(m, e[i]);
    sm[threadIdx.x] = m;
    __syncthreads();
    for (int s = 512; s > 0; s >>= 1) {
        if ((int)threadIdx.x < s) sm[threadIdx.x] = fmaxf(sm[threadIdx.x], sm[threadIdx.x + s]);
        __syncthreads();
    }
    if (threadIdx.x == 0) *emax = sm[0];
}

// ---------- masked softmax over T; writes a (bf16 + f32) and attentions ----------
__global__ void attn_softmax_k(const float* __restrict__ e, const float* __restrict__ emax,
                               const int* __restrict__ seqlens, bf16_t* __restrict__ a_bf,
                               float* __restrict__ a_f32,
                               float* __restrict__ att_out, int step) {
    int b = blockIdx.x;
    int tid = threadIdx.x;
    __shared__ float sm[256];
    float gm = *emax;
    int len = seqlens[b];
    float ex[4];
    float partial = 0.0f;
#pragma unroll
    for (int c = 0; c < 4; ++c) {
        int t = c * 256 + tid;
        float v = __expf(e[(size_t)b * T_ + t] - gm) * ((t < len) ? 1.0f : 0.0f);
        ex[c] = v;
        partial += v;
    }
    sm[tid] = partial;
    __syncthreads();
    for (int s = 128; s > 0; s >>= 1) {
        if (tid < s) sm[tid] += sm[tid + s];
        __syncthreads();
    }
    float denom = sm[0] + EPS_;
#pragma unroll
    for (int c = 0; c < 4; ++c) {
        int t = c * 256 + tid;
        float av = ex[c] / denom;
        a_bf[(size_t)b * T_ + t] = f2bf(av);
        a_f32[(size_t)b * T_ + t] = av;
        att_out[((size_t)b * L_ + step) * T_ + t] = av;
    }
}

// ---------- context via WMMA: ctx[b,:] = a[b,:] @ h[b]  (a replicated on M) ----------
__global__ void attn_ctx_wmma_k(const bf16_t* __restrict__ a_bf,
                                const bf16_t* __restrict__ hBpack,
                                bf16_t* __restrict__ Xbf, bf16_t* __restrict__ featbf) {
    int wave = (blockIdx.x * blockDim.x + threadIdx.x) >> 5;
    if (wave >= B_ * 8) return;           // 64 b * (32 ntiles / 4)
    int l = threadIdx.x & 31;
    int b = wave >> 3;
    int nb = wave & 7;
    int lhalf = (l >> 4) & 1;
    const bf16_t* arow = a_bf + (size_t)b * T_;
    v8f acc0 = {}, acc1 = {}, acc2 = {}, acc3 = {};
    for (int kt = 0; kt < 32; ++kt) {
        int k0 = kt * 32 + lhalf * 8;
        v16bf a = load_a_frag(arow, k0);  // a replicated across all 16 M rows
        const bf16_t* bp = hBpack + ((((size_t)b * 32 + kt) * 32 + nb * 4) * 32 + l) * 16;
        __builtin_prefetch(bp + 32 * 512, 0, 1);   // next kt block
        v16bf b0 = load_pk_frag(bp);
        v16bf b1 = load_pk_frag(bp + 512);
        v16bf b2 = load_pk_frag(bp + 1024);
        v16bf b3 = load_pk_frag(bp + 1536);
        acc0 = __builtin_amdgcn_wmma_f32_16x16x32_bf16(false, a, false, b0, (short)0, acc0, false, false);
        acc1 = __builtin_amdgcn_wmma_f32_16x16x32_bf16(false, a, false, b1, (short)0, acc1, false, false);
        acc2 = __builtin_amdgcn_wmma_f32_16x16x32_bf16(false, a, false, b2, (short)0, acc2, false, false);
        acc3 = __builtin_amdgcn_wmma_f32_16x16x32_bf16(false, a, false, b3, (short)0, acc3, false, false);
    }
    if (l < 16) {  // row M=0 of each tile; lanes 0..15 give N=0..15
#pragma unroll
        for (int q = 0; q < 4; ++q) {
            float v = (q == 0) ? acc0[0] : (q == 1) ? acc1[0] : (q == 2) ? acc2[0] : acc3[0];
            int c = (nb * 4 + q) * 16 + l;
            bf16_t cb = f2bf(v);
            Xbf[(size_t)b * KX_ + V_ + c] = cb;       // x_next[:, 64:576] = ctx
            featbf[(size_t)b * 1024 + 512 + c] = cb;  // feat[:, 512:] = ctx
        }
    }
}

// ---------- fallback context (fp32 VALU) if workspace is too small ----------
__global__ void attn_ctx_valu_k(const float* __restrict__ a, const float* __restrict__ h,
                                bf16_t* __restrict__ Xbf, bf16_t* __restrict__ featbf) {
    int idx = blockIdx.x * blockDim.x + threadIdx.x;
    if (idx >= B_ * HC_) return;
    int b = idx >> 9, c = idx & 511;
    const float* arow = a + (size_t)b * T_;
    const float* hb = h + (size_t)b * T_ * HC_ + c;
    float s = 0.0f;
    for (int t = 0; t < T_; t += 4) {
        if ((t & 63) == 0) __builtin_prefetch(hb + (size_t)(t + 64) * HC_, 0, 1);
        s = fmaf(arow[t + 0], hb[(size_t)(t + 0) * HC_], s);
        s = fmaf(arow[t + 1], hb[(size_t)(t + 1) * HC_], s);
        s = fmaf(arow[t + 2], hb[(size_t)(t + 2) * HC_], s);
        s = fmaf(arow[t + 3], hb[(size_t)(t + 3) * HC_], s);
    }
    bf16_t cb = f2bf(s);
    Xbf[(size_t)b * KX_ + V_ + c] = cb;
    featbf[(size_t)b * 1024 + 512 + c] = cb;
}

// ---------- per-row softmax over V=64, argmax/EOS, emit y_hat ----------
__global__ void finalize_k(const float* __restrict__ logits, float* __restrict__ yh_out,
                           int* __restrict__ lens, bf16_t* __restrict__ Xbf, int step) {
    int b = blockIdx.x, v = threadIdx.x;
    __shared__ float sf[64];
    __shared__ int si[64];
    float x = logits[b * 64 + v];
    sf[v] = x;
    __syncthreads();
    for (int s = 32; s > 0; s >>= 1) {
        if (v < s) sf[v] = fmaxf(sf[v], sf[v + s]);
        __syncthreads();
    }
    float m = sf[0];
    __syncthreads();
    float ex = __expf(x - m);
    sf[v] = ex;
    __syncthreads();
    for (int s = 32; s > 0; s >>= 1) {
        if (v < s) sf[v] += sf[v + s];
        __syncthreads();
    }
    float y = ex / sf[0];
    __syncthreads();
    yh_out[((size_t)b * L_ + step) * V_ + v] = y;
    Xbf[(size_t)b * KX_ + v] = f2bf(y);
    sf[v] = y; si[v] = v;
    __syncthreads();
    for (int s = 32; s > 0; s >>= 1) {
        if (v < s) {
            if (sf[v + s] > sf[v]) { sf[v] = sf[v + s]; si[v] = si[v + s]; }
        }
        __syncthreads();
    }
    if (v == 0) {
        if (si[0] == 1 /*EOS*/ && lens[b] > step) lens[b] = step + 1;
    }
}

__global__ void write_lens_k(const int* __restrict__ lens, int* __restrict__ out) {
    int b = threadIdx.x;
    if (b < B_) out[b] = lens[b];
}

// ---------- host launcher ----------
extern "C" void kernel_launch(void* const* d_in, const int* in_sizes, int n_in,
                              void* d_out, int out_size, void* d_ws, size_t ws_size,
                              hipStream_t stream) {
    const float* h     = (const float*)d_in[0];
    const float* W_ih  = (const float*)d_in[1];
    const float* W_hh  = (const float*)d_in[2];
    const float* b_ih  = (const float*)d_in[3];
    const float* b_hh  = (const float*)d_in[4];
    const float* fc1_w = (const float*)d_in[5];
    const float* fc1_b = (const float*)d_in[6];
    const float* fc2_w = (const float*)d_in[7];
    const int*   slens = (const int*)d_in[8];

    char* ws = (char*)d_ws;
    size_t off = 0;
    auto alloc = [&](size_t bytes) -> void* {
        void* p = ws + off;
        off += (bytes + 255) & ~(size_t)255;
        return p;
    };
    bf16_t* hbf     = (bf16_t*)alloc((size_t)B_ * T_ * HC_ * 2);   // 67 MB (scores A-layout)
    bf16_t* Wpack   = (bf16_t*)alloc((size_t)34 * 128 * 512 * 2);
    bf16_t* fc1pack = (bf16_t*)alloc((size_t)32 * 8 * 512 * 2);
    bf16_t* fc2pack = (bf16_t*)alloc((size_t)4 * 4 * 512 * 2);
    bf16_t* Xbf     = (bf16_t*)alloc((size_t)B_ * KX_ * 2);        // [y_hat|ctx|hs]
    bf16_t* featbf  = (bf16_t*)alloc((size_t)B_ * 1024 * 2);       // [hs|ctx]
    bf16_t* h1bf    = (bf16_t*)alloc((size_t)B_ * HM_ * 2);
    float*  gates   = (float*)alloc((size_t)B_ * G4_ * 4);
    float*  cs      = (float*)alloc((size_t)B_ * HS_ * 4);
    float*  e       = (float*)alloc((size_t)B_ * T_ * 4);
    bf16_t* a_bf    = (bf16_t*)alloc((size_t)B_ * T_ * 2);
    float*  a_f32   = (float*)alloc((size_t)B_ * T_ * 4);
    float*  bsum    = (float*)alloc((size_t)G4_ * 4);
    float*  emax    = (float*)alloc(256);
    int*    lens    = (int*)alloc((size_t)B_ * 4);
    float*  logits  = (float*)alloc((size_t)B_ * V_ * 4);
    bf16_t* hBpack  = (bf16_t*)alloc((size_t)B_ * T_ * HC_ * 2);   // 67 MB (ctx B-layout)
    bool use_wmma_ctx = (off <= ws_size);
    (void)in_sizes; (void)n_in; (void)out_size;

    float* yh_out  = (float*)d_out;                     // [64,128,64]
    int*   len_out = (int*)((float*)d_out + B_ * L_ * V_);
    float* att_out = (float*)d_out + B_ * L_ * V_ + B_; // [64,1,128,1024]

    // --- prep (once per call) ---
    conv_h_bf16_k<<<(B_ * T_ * HC_) / 256, 256, 0, stream>>>(h, hbf);
    if (use_wmma_ctx)
        pack_hB_k<<<(B_ * T_ * HC_) / 256, 256, 0, stream>>>(h, hBpack);
    pack_gates_w_k<<<(34 * 128 * 512 + 255) / 256, 256, 0, stream>>>(W_ih, W_hh, b_ih, b_hh, Wpack, bsum);
    pack_fc_k<<<(32 * 8 * 512 + 4 * 4 * 512 + 255) / 256, 256, 0, stream>>>(fc1_w, fc2_w, fc1pack, fc2pack);
    init_state_k<<<(B_ * KX_ + 255) / 256, 256, 0, stream>>>(h, Xbf, cs, lens, featbf);

    // --- 128 sequential decode steps ---
    for (int step = 0; step < L_; ++step) {
        // gates = X[64,1088] @ Wcat[1088,2048] + (b_ih+b_hh): 4 mtiles * 32 nblocks = 128 waves
        wmma_gemm_k<<<16, 256, 0, stream>>>(Xbf, KX_, Wpack, 128, 34, 4,
                                            gates, G4_, bsum, nullptr, 0);
        lstm_cell_k<<<(B_ * HS_) / 256, 256, 0, stream>>>(gates, cs, Xbf, featbf);
        attn_scores_k<<<(B_ * 64 * 32) / 256, 256, 0, stream>>>(hbf, featbf, e);
        reduce_max_k<<<1, 1024, 0, stream>>>(e, emax);
        attn_softmax_k<<<B_, 256, 0, stream>>>(e, emax, slens, a_bf, a_f32, att_out, step);
        if (use_wmma_ctx)
            attn_ctx_wmma_k<<<(B_ * 8 * 32) / 256, 256, 0, stream>>>(a_bf, hBpack, Xbf, featbf);
        else
            attn_ctx_valu_k<<<(B_ * HC_) / 256, 256, 0, stream>>>(a_f32, h, Xbf, featbf);
        // h1 = feat[64,1024] @ fc1T[1024,128] + fc1_b -> bf16 : 4 m * 2 nb = 8 waves
        wmma_gemm_k<<<1, 256, 0, stream>>>(featbf, 1024, fc1pack, 8, 32, 4,
                                           nullptr, 0, fc1_b, h1bf, HM_);
        // logits = h1[64,128] @ fc2T[128,64] : 4 m * 1 nb = 4 waves
        wmma_gemm_k<<<1, 256, 0, stream>>>(h1bf, HM_, fc2pack, 4, 4, 4,
                                           logits, V_, nullptr, nullptr, 0);
        finalize_k<<<B_, 64, 0, stream>>>(logits, yh_out, lens, Xbf, step);
    }
    write_lens_k<<<1, 64, 0, stream>>>(lens, len_out);
}
